// GRUModel_34127810134418
// MI455X (gfx1250) — matrix operand order repaired and checked by reading, not compiled, not run on hardware
//
#include <hip/hip_runtime.h>
#include <hip/hip_bf16.h>

typedef __attribute__((ext_vector_type(16))) _Float16 v16h;
typedef __attribute__((ext_vector_type(8)))  float    v8f;

#define BSZ  8192
#define TLEN 1024
#define ISZ  2
#define HSZ  10
#define OSZ  2

// ---- division-free activations -------------------------------------------------
__device__ __forceinline__ float fast_tanh(float x) {
#if __has_builtin(__builtin_amdgcn_tanhf)
    return __builtin_amdgcn_tanhf(x);          // v_tanh_f32 (gfx1250)
#else
    float ax = __builtin_fabsf(x);
    float e  = __expf(-2.0f * ax);             // in (0,1], never overflows
    float t  = (1.0f - e) * __builtin_amdgcn_rcpf(1.0f + e);   // v_rcp_f32, no div
    return __builtin_copysignf(t, x);
#endif
}
__device__ __forceinline__ float fast_sigmoid(float x) {
#if __has_builtin(__builtin_amdgcn_tanhf)
    return __builtin_fmaf(0.5f, __builtin_amdgcn_tanhf(0.5f * x), 0.5f);
#else
    return __builtin_amdgcn_rcpf(1.0f + __expf(-x));
#endif
}

// Row semantics inside a 3-tile (48-row) fused gate matrix:
//   tile t, row M (0..15):  kind = M&3 (0=r,1=z,2=hn,3=xn)
//   j = ((M>>2)&1) + 2*t + (M>=8 ? 5 : 0) ; rows with (t==2 && (M&4)) are padding.
// With the documented D layout (lane L<16 holds M=0..7, lane L+16 holds M=8..15),
// lane L owns (r,z,hn,xn) for j=0..4 of batch column L; lane L+16 owns j=5..9.

// A-matrix 16x32 f16 fragment, contiguous K as used here (B matches): lane L<16 holds row M=L,
// K0..15; lane L+16 holds row M=L, K16..31.
__device__ __forceinline__ v16h build_a_frag(const float* __restrict__ Wih,
                                             const float* __restrict__ Whh,
                                             int xdim, int tile, int lane) {
    const int M    = lane & 15;
    const int ksel = lane >> 4;
    const bool pad = (tile == 2) && (M & 4);
    const int kind = M & 3;
    const int j    = ((M >> 2) & 1) + 2 * tile + ((M >= 8) ? 5 : 0);
    const int row  = (kind == 0) ? j : (kind == 1) ? (HSZ + j) : (2 * HSZ + j);
    // hn rows (kind==2) have no x-side columns; xn rows (kind==3) have no h-side.
    const float xm = (pad || kind == 2) ? 0.0f : 1.0f;
    const float hm = (pad || kind == 3) ? 0.0f : 1.0f;
    const int rsafe = pad ? 0 : row;     // keep addresses in-bounds for pad rows

    v16h a;
#pragma unroll
    for (int e = 0; e < 16; ++e) {
        const int K = e + 16 * ksel;   // absolute K column
        float v = 0.0f;
        if (K < xdim)                 v = Wih[rsafe * xdim + K] * xm;
        else if (K < xdim + HSZ)      v = Whh[rsafe * HSZ + (K - xdim)] * hm;
        a[e] = (_Float16)v;
    }
    return a;
}

// C/D 16x16 f32 layout: lane L<16 -> vgpr v = (M=v, N=L); lane L+16 -> (M=v+8, N=L-16).
__device__ __forceinline__ v8f build_bias_frag(const float* __restrict__ bih,
                                               const float* __restrict__ bhh,
                                               int tile, int lane) {
    const int mbase = (lane < 16) ? 0 : 8;
    v8f c;
#pragma unroll
    for (int v = 0; v < 8; ++v) {
        const int M    = mbase + v;
        const bool pad = (tile == 2) && (M & 4);
        const int kind = M & 3;
        const int j    = ((M >> 2) & 1) + 2 * tile + ((M >= 8) ? 5 : 0);
        const int row  = (kind == 0) ? j : (kind == 1) ? (HSZ + j) : (2 * HSZ + j);
        const int rs   = pad ? 0 : row;
        float b = (kind == 2) ? bhh[rs]
                : (kind == 3) ? bih[rs]
                              : (bih[rs] + bhh[rs]);
        c[v] = pad ? 0.0f : b;
    }
    return c;
}

#define WMMA_F16(A, Bf, C) \
    __builtin_amdgcn_wmma_f32_16x16x32_f16(false, (A), false, (Bf), (short)0, (C), false, false)

// One gate quadruple -> new hidden value.  d[4l+0..3] = (r,z,hn,xn) pre-activations.
// All indices compile-time constant: hOwn/hnew are fixed 5-element register arrays.
#define GATE_STEP(dreg, l, jl, hOwn, hnew)                                 \
    {                                                                      \
        float pre_r  = (dreg)[4 * (l) + 0];                                \
        float pre_z  = (dreg)[4 * (l) + 1];                                \
        float pre_hn = (dreg)[4 * (l) + 2];                                \
        float pre_xn = (dreg)[4 * (l) + 3];                                \
        float r = fast_sigmoid(pre_r);                                     \
        float z = fast_sigmoid(pre_z);                                     \
        float n = fast_tanh(__builtin_fmaf(r, pre_hn, pre_xn));            \
        (hnew)[(jl)] = __builtin_fmaf(z, (hOwn)[(jl)] - n, n);             \
    }

__global__ __launch_bounds__(32) void gru2_fused_wmma(
    const float* __restrict__ X,
    const float* __restrict__ Wih0, const float* __restrict__ Whh0,
    const float* __restrict__ bih0, const float* __restrict__ bhh0,
    const float* __restrict__ Wih1, const float* __restrict__ Whh1,
    const float* __restrict__ bih1, const float* __restrict__ bhh1,
    const float* __restrict__ Wfc,  const float* __restrict__ bfc,
    float* __restrict__ out)
{
    const int lane = threadIdx.x;   // wave32: one wave per block
    const int half = lane >> 4;     // 0 -> owns j=0..4, 1 -> owns j=5..9
    const int col  = lane & 15;     // batch column within this 16-wide tile
    const int b0   = blockIdx.x * 16;

    // ---- one-time setup: weight A-fragments and bias C-fragments (loop-invariant)
    v16h A0[3], A1[3];
    v8f  C0[3], C1[3];
#pragma unroll
    for (int tl = 0; tl < 3; ++tl) {
        A0[tl] = build_a_frag(Wih0, Whh0, ISZ, tl, lane);
        A1[tl] = build_a_frag(Wih1, Whh1, HSZ, tl, lane);
        C0[tl] = build_bias_frag(bih0, bhh0, tl, lane);
        C1[tl] = build_bias_frag(bih1, bhh1, tl, lane);
    }

    // Hidden state, split: hOwn = this lane's 5 units, hOth = partner lane's 5.
    // (half 0: own = abs j 0..4, oth = abs j 5..9;  half 1: swapped.)
    float h0own[5], h0oth[5], h1own[5], h1oth[5];
#pragma unroll
    for (int i = 0; i < 5; ++i) { h0own[i] = 0.f; h0oth[i] = 0.f; h1own[i] = 0.f; h1oth[i] = 0.f; }

    const float* __restrict__ xp = X + (size_t)(b0 + col) * (TLEN * ISZ);

    for (int t = 0; t < TLEN; ++t) {
        if ((t & 15) == 0 && t + 32 < TLEN)
            __builtin_prefetch(xp + (t + 32) * ISZ, 0, 0);   // global_prefetch_b8

        float2 xv = *(const float2*)(xp + t * ISZ);

        // ---------------- layer 0:  K-vector = [x_t(2) | h0(10)], K=12 ----------------
        v16h bf0;
        if (half == 0) {
            bf0[0] = (_Float16)xv.x;
            bf0[1] = (_Float16)xv.y;
#pragma unroll
            for (int i = 0; i < 5; ++i) {
                bf0[2 + i] = (_Float16)h0own[i];   // abs j = 0..4
                bf0[7 + i] = (_Float16)h0oth[i];   // abs j = 5..9
            }
#pragma unroll
            for (int e = 12; e < 16; ++e) bf0[e] = (_Float16)0.0f;
        } else {
#pragma unroll
            for (int e = 0; e < 16; ++e) bf0[e] = (_Float16)0.0f;   // K16..31 = 0
        }

        v8f d0 = WMMA_F16(A0[0], bf0, C0[0]);
        v8f d1 = WMMA_F16(A0[1], bf0, C0[1]);
        v8f d2 = WMMA_F16(A0[2], bf0, C0[2]);

        float hn0[5];
        GATE_STEP(d0, 0, 0, h0own, hn0)
        GATE_STEP(d0, 1, 1, h0own, hn0)
        GATE_STEP(d1, 0, 2, h0own, hn0)
        GATE_STEP(d1, 1, 3, h0own, hn0)
        GATE_STEP(d2, 0, 4, h0own, hn0)
#pragma unroll
        for (int i = 0; i < 5; ++i) {
            h0oth[i] = __shfl_xor(hn0[i], 16, 32);
            h0own[i] = hn0[i];
        }

        // ---------------- layer 1:  K-vector = [h0_new(10) | h1(10)], K=20 ------------
        v16h bf1;
        if (half == 0) {
#pragma unroll
            for (int i = 0; i < 5; ++i) {
                bf1[i]     = (_Float16)h0own[i];   // h0 abs 0..4
                bf1[5 + i] = (_Float16)h0oth[i];   // h0 abs 5..9
            }
#pragma unroll
            for (int i = 0; i < 5; ++i)
                bf1[10 + i] = (_Float16)h1own[i];  // h1 abs 0..4
            bf1[15] = (_Float16)h1oth[0];          // h1 abs 5
        } else {
            // lane L+16 supplies K16..31 of column L: h1 abs 6..9 = own[1..4]
            bf1[0] = (_Float16)h1own[1];
            bf1[1] = (_Float16)h1own[2];
            bf1[2] = (_Float16)h1own[3];
            bf1[3] = (_Float16)h1own[4];
#pragma unroll
            for (int e = 4; e < 16; ++e) bf1[e] = (_Float16)0.0f;   // K20..31 = 0
        }

        v8f e0 = WMMA_F16(A1[0], bf1, C1[0]);
        v8f e1 = WMMA_F16(A1[1], bf1, C1[1]);
        v8f e2 = WMMA_F16(A1[2], bf1, C1[2]);

        float hn1[5];
        GATE_STEP(e0, 0, 0, h1own, hn1)
        GATE_STEP(e0, 1, 1, h1own, hn1)
        GATE_STEP(e1, 0, 2, h1own, hn1)
        GATE_STEP(e1, 1, 3, h1own, hn1)
        GATE_STEP(e2, 0, 4, h1own, hn1)
#pragma unroll
        for (int i = 0; i < 5; ++i) {
            h1oth[i] = __shfl_xor(hn1[i], 16, 32);
            h1own[i] = hn1[i];
        }
    }

    // ---------------- FC head on final h1 ----------------
    if (half == 0) {      // own = abs j 0..4, oth = abs j 5..9
        float acc0 = bfc[0], acc1 = bfc[1];
#pragma unroll
        for (int i = 0; i < 5; ++i) {
            acc0 = __builtin_fmaf(h1own[i], Wfc[i],           acc0);
            acc0 = __builtin_fmaf(h1oth[i], Wfc[5 + i],       acc0);
            acc1 = __builtin_fmaf(h1own[i], Wfc[HSZ + i],     acc1);
            acc1 = __builtin_fmaf(h1oth[i], Wfc[HSZ + 5 + i], acc1);
        }
        float* op = out + (size_t)(b0 + col) * OSZ;
        op[0] = acc0;
        op[1] = acc1;
    }
}

extern "C" void kernel_launch(void* const* d_in, const int* in_sizes, int n_in,
                              void* d_out, int out_size, void* d_ws, size_t ws_size,
                              hipStream_t stream) {
    (void)in_sizes; (void)n_in; (void)d_ws; (void)ws_size; (void)out_size;
    const float* X    = (const float*)d_in[0];
    const float* Wih0 = (const float*)d_in[1];
    const float* Whh0 = (const float*)d_in[2];
    const float* bih0 = (const float*)d_in[3];
    const float* bhh0 = (const float*)d_in[4];
    const float* Wih1 = (const float*)d_in[5];
    const float* Whh1 = (const float*)d_in[6];
    const float* bih1 = (const float*)d_in[7];
    const float* bhh1 = (const float*)d_in[8];
    const float* Wfc  = (const float*)d_in[9];
    const float* bfc  = (const float*)d_in[10];
    float* out = (float*)d_out;

    dim3 grid(BSZ / 16);   // 512 independent 16-batch tiles, one wave each
    dim3 block(32);
    gru2_fused_wmma<<<grid, block, 0, stream>>>(X, Wih0, Whh0, bih0, bhh0,
                                                Wih1, Whh1, bih1, bhh1,
                                                Wfc, bfc, out);
}
